// SpatialGatingUnit_35897336660176
// MI455X (gfx1250) — compile-verified, matrix operand order checked by reading, not run on hardware
//
#include <hip/hip_runtime.h>
#include <hip/hip_bf16.h>

typedef __attribute__((ext_vector_type(16))) __bf16 v16bf;
typedef __attribute__((ext_vector_type(8)))  __bf16 v8bf;
typedef __attribute__((ext_vector_type(4)))  __bf16 v4bf;
typedef __attribute__((ext_vector_type(8)))  float  v8f;

#define D_FEAT 128

static __device__ __forceinline__ __bf16 f2bf(float f) {
    unsigned u = __builtin_bit_cast(unsigned, f);
    unsigned r = (u + 0x7FFFu + ((u >> 16) & 1u)) >> 16;
    unsigned short s = (unsigned short)r;
    return __builtin_bit_cast(__bf16, s);
}

static __device__ __forceinline__ float wave_sum32(float v) {
    for (int off = 16; off > 0; off >>= 1)
        v += __shfl_xor(v, off, 32);
    return v;
}

// ---- degree init: deg[i] = 2.0 (improved self-loop weight) ----
__global__ void k_init_deg(float* deg, int N) {
    int i = blockIdx.x * blockDim.x + threadIdx.x;
    if (i < N) deg[i] = 2.0f;
}

// ---- deg[col[e]] += 1.0 for real edges ----
__global__ void k_deg(const int* col, float* deg, int E) {
    int e = blockIdx.x * blockDim.x + threadIdx.x;
    if (e < E) atomicAdd(&deg[col[e]], 1.0f);
}

// ---- dinv = deg > 0 ? rsqrt(max(deg,1e-12)) : 0 ----
__global__ void k_dinv(const float* deg, float* dinv, int N) {
    int i = blockIdx.x * blockDim.x + threadIdx.x;
    if (i < N) {
        float d = deg[i];
        dinv[i] = (d > 0.0f) ? rsqrtf(fmaxf(d, 1e-12f)) : 0.0f;
    }
}

// ---- W (f32 row-major [out,in]) -> bf16 ----
__global__ void k_wconv(const float* W, __bf16* Wb, int n) {
    int i = blockIdx.x * blockDim.x + threadIdx.x;
    if (i < n) Wb[i] = f2bf(W[i]);
}

// ---- LayerNorm: one wave32 per row, emit bf16 gate ----
__global__ __launch_bounds__(256) void k_ln(const float* __restrict__ x,
                                            const float* __restrict__ gamma,
                                            const float* __restrict__ beta,
                                            __bf16* __restrict__ gate, int N) {
    int wave = (blockIdx.x * blockDim.x + threadIdx.x) >> 5;
    int lane = threadIdx.x & 31;
    if (wave >= N) return;
    const float* xr = x + (size_t)wave * D_FEAT;
    float4 v = *(const float4*)(xr + lane * 4);
    float s  = v.x + v.y + v.z + v.w;
    float s2 = v.x * v.x + v.y * v.y + v.z * v.z + v.w * v.w;
    s  = wave_sum32(s);
    s2 = wave_sum32(s2);
    float mu  = s * (1.0f / D_FEAT);
    float var = s2 * (1.0f / D_FEAT) - mu * mu;
    float rs  = rsqrtf(var + 1e-5f);
    float4 g4 = *(const float4*)(gamma + lane * 4);
    float4 b4 = *(const float4*)(beta + lane * 4);
    v4bf o;
    o[0] = f2bf((v.x - mu) * rs * g4.x + b4.x);
    o[1] = f2bf((v.y - mu) * rs * g4.y + b4.y);
    o[2] = f2bf((v.z - mu) * rs * g4.z + b4.z);
    o[3] = f2bf((v.w - mu) * rs * g4.w + b4.w);
    *(v4bf*)(gate + (size_t)wave * D_FEAT + lane * 4) = o;
}

// ---- WMMA GEMM: h = gate @ W^T ; agg init = 2*dinv^2*h (self-loop term) ----
__global__ __launch_bounds__(256) void k_gemm(const __bf16* __restrict__ gate,
                                              const __bf16* __restrict__ Wb,
                                              const float* __restrict__ dinv,
                                              float* __restrict__ h,
                                              float* __restrict__ agg, int N) {
    int wave = threadIdx.x >> 5;        // 0..7 : output column tile
    int lane = threadIdx.x & 31;
    int m0 = blockIdx.x * 16;
    int n0 = wave * 16;
    int lr   = lane & 15;
    int koff = (lane >> 4) * 8;

    int ra = m0 + lr;
    if (ra >= N) ra = N - 1;            // clamp reads; EXEC must stay all-1 for WMMA
    const __bf16* arow = gate + (size_t)ra * D_FEAT;
    const __bf16* brow = Wb   + (size_t)(n0 + lr) * D_FEAT;

    v8f c = {};
    #pragma unroll
    for (int kc = 0; kc < D_FEAT; kc += 32) {
        v8bf alo = *(const v8bf*)(arow + kc + koff);
        v8bf ahi = *(const v8bf*)(arow + kc + 16 + koff);
        v8bf blo = *(const v8bf*)(brow + kc + koff);
        v8bf bhi = *(const v8bf*)(brow + kc + 16 + koff);
        v16bf a = __builtin_shufflevector(alo, ahi, 0,1,2,3,4,5,6,7,8,9,10,11,12,13,14,15);
        v16bf b = __builtin_shufflevector(blo, bhi, 0,1,2,3,4,5,6,7,8,9,10,11,12,13,14,15);
        c = __builtin_amdgcn_wmma_f32_16x16x32_bf16(false, a, false, b, (short)0, c, false, false);
    }

    int n  = n0 + lr;
    int mb = m0 + ((lane >> 4) << 3);
    if (m0 + 16 <= N) {
        // full tile (all blocks except possibly the last): straight-line stores
        #pragma unroll
        for (int v = 0; v < 8; ++v) {
            int m = mb + v;
            float hv = c[v];
            float dv = dinv[m];
            h[(size_t)m * D_FEAT + n]   = hv;
            agg[(size_t)m * D_FEAT + n] = hv * (2.0f * dv * dv);
        }
    } else {
        #pragma unroll
        for (int v = 0; v < 8; ++v) {
            int m = mb + v;
            if (m < N) {
                float hv = c[v];
                float dv = dinv[m];
                h[(size_t)m * D_FEAT + n]   = hv;
                agg[(size_t)m * D_FEAT + n] = hv * (2.0f * dv * dv);
            }
        }
    }
}

// ---- edge scatter: one wave per edge, 4 float atomics per lane ----
__global__ __launch_bounds__(256) void k_scatter(const int* __restrict__ rows,
                                                 const int* __restrict__ cols,
                                                 const float* __restrict__ h,
                                                 const float* __restrict__ dinv,
                                                 float* __restrict__ agg, int E) {
    int e = blockIdx.x * 8 + (threadIdx.x >> 5);
    if (e >= E) return;
    int lane = threadIdx.x & 31;
    int r = rows[e];
    int c = cols[e];
    float norm = dinv[r] * dinv[c];
    float4 hv = *(const float4*)(h + (size_t)r * D_FEAT + lane * 4);
    float* dst = agg + (size_t)c * D_FEAT + lane * 4;
    atomicAdd(dst + 0, hv.x * norm);
    atomicAdd(dst + 1, hv.y * norm);
    atomicAdd(dst + 2, hv.z * norm);
    atomicAdd(dst + 3, hv.w * norm);
}

// ---- epilogue: out = tanh(agg + b) * x (in place on d_out) ----
__global__ void k_final(float* __restrict__ agg, const float* __restrict__ x,
                        const float* __restrict__ b, long total) {
    long i = (long)blockIdx.x * blockDim.x + threadIdx.x;
    if (i < total) {
        float a = agg[i] + b[i & (D_FEAT - 1)];
        agg[i] = tanhf(a) * x[i];
    }
}

extern "C" void kernel_launch(void* const* d_in, const int* in_sizes, int n_in,
                              void* d_out, int out_size, void* d_ws, size_t ws_size,
                              hipStream_t stream) {
    const float* x     = (const float*)d_in[0];
    const int*   ei    = (const int*)  d_in[1];
    const float* gamma = (const float*)d_in[2];
    const float* beta  = (const float*)d_in[3];
    const float* W     = (const float*)d_in[4];
    const float* bias  = (const float*)d_in[5];
    float* out = (float*)d_out;

    const int N = in_sizes[0] / D_FEAT;
    const int E = in_sizes[1] / 2;
    const int* rows = ei;
    const int* cols = ei + E;

    // workspace layout (all offsets 256B-aligned)
    char* ws = (char*)d_ws;
    size_t off = 0;
    __bf16* gate = (__bf16*)(ws + off); off += ((size_t)N * D_FEAT * 2 + 255) & ~(size_t)255;
    float*  h    = (float*) (ws + off); off += ((size_t)N * D_FEAT * 4 + 255) & ~(size_t)255;
    __bf16* Wb   = (__bf16*)(ws + off); off += ((size_t)D_FEAT * D_FEAT * 2 + 255) & ~(size_t)255;
    float*  deg  = (float*) (ws + off); off += ((size_t)N * 4 + 255) & ~(size_t)255;
    float*  dinv = (float*) (ws + off); off += ((size_t)N * 4 + 255) & ~(size_t)255;

    const int B = 256;
    k_init_deg<<<(N + B - 1) / B, B, 0, stream>>>(deg, N);
    k_deg<<<(E + B - 1) / B, B, 0, stream>>>(cols, deg, E);
    k_dinv<<<(N + B - 1) / B, B, 0, stream>>>(deg, dinv, N);
    k_wconv<<<(D_FEAT * D_FEAT + B - 1) / B, B, 0, stream>>>(W, Wb, D_FEAT * D_FEAT);
    k_ln<<<(N * 32 + B - 1) / B, B, 0, stream>>>(x, gamma, beta, gate, N);
    k_gemm<<<(N + 15) / 16, B, 0, stream>>>(gate, Wb, dinv, h, out, N);
    k_scatter<<<(E + 7) / 8, B, 0, stream>>>(rows, cols, h, dinv, out, E);
    long total = (long)N * D_FEAT;
    k_final<<<(int)((total + B - 1) / B), B, 0, stream>>>(out, x, bias, total);
}
